// LiteMLA_39840116638163
// MI455X (gfx1250) — matrix operand order repaired and checked by reading, not compiled, not run on hardware
//
#include <hip/hip_runtime.h>

typedef __attribute__((ext_vector_type(16))) _Float16 v16h;
typedef __attribute__((ext_vector_type(8)))  _Float16 v8h;
typedef __attribute__((ext_vector_type(8)))  float    v8f;

#define HW    4096
#define WDIM  64
#define TD    768     // 3 * HEADS * DIM
#define CIN   256
#define COUT  256
#define CMID  512     // 2 * HEADS * DIM
#define NG    64      // attention groups
#define NB    8

// ---------------- workspace layout (bytes) ----------------
#define OFF_WQKV   ((size_t)0)                          // 768*256 f16
#define OFF_WPROJ  (OFF_WQKV  + (size_t)768*256*2)      // 256*512 f16
#define OFF_SCALE  (OFF_WPROJ + (size_t)256*512*2)      // 256 f32
#define OFF_SHIFT  (OFF_SCALE + (size_t)256*4)          // 256 f32
#define OFF_XT     (OFF_SHIFT + (size_t)256*4)          // tiled x   f16: 8*256*4096
#define OFF_QKV    (OFF_XT    + (size_t)NB*CIN*HW*2)    // qkv       f16: 8*768*4096
#define OFF_AGG    (OFF_QKV   + (size_t)NB*TD*HW*2)     // agg       f16: 8*768*4096
#define OFF_ATTN   (OFF_AGG   + (size_t)NB*TD*HW*2)     // tiled attn f16: 8*512*4096

// Tiled ("fragment-major") activation layout for WMMA B operands:
//   tile index = ((b*KBLK + kk)*256 + ntile), each tile = 512 halves laid out
//   [lane(32)][j(16)] with lane = (k_in_32 >= 16)*16 + (n&15), j = k_in_32 & 15.
// A lane's 16 halves for one 16x16x32 WMMA are 32 contiguous bytes.

// ---------------- K0: weight convert + BN prep ----------------
__global__ void prep_kernel(const float* __restrict__ wqkv,
                            const float* __restrict__ wproj,
                            const float* __restrict__ gamma,
                            const float* __restrict__ beta,
                            const float* __restrict__ mean,
                            const float* __restrict__ var,
                            _Float16* __restrict__ wqkv16,
                            _Float16* __restrict__ wproj16,
                            float* __restrict__ scale,
                            float* __restrict__ shift) {
  const int idx = blockIdx.x * blockDim.x + threadIdx.x;
  if (idx < 768 * 256)  wqkv16[idx]  = (_Float16)wqkv[idx];
  if (idx < 256 * 512)  wproj16[idx] = (_Float16)wproj[idx];
  if (idx < 256) {
    float s = gamma[idx] * rsqrtf(var[idx] + 1e-5f);
    scale[idx] = s;
    shift[idx] = beta[idx] - mean[idx] * s;
  }
}

// ---------------- K0b: tile x (f32 -> f16 fragment-major) ----------------
// grid (32, 8, 8), block 256: thread -> (b, kk, ntile, lane); writes 32B chunk.
__global__ void xtile_kernel(const float* __restrict__ x,
                             _Float16* __restrict__ xt) {
  const int lane  = threadIdx.x & 31;
  const int sub   = threadIdx.x >> 5;
  const int ntile = blockIdx.x * 8 + sub;
  const int kk    = blockIdx.y;
  const int b     = blockIdx.z;
  const int n     = ntile * 16 + (lane & 15);
  const int kbase = kk * 32 + ((lane >> 4) << 4);

  const float* src = x + ((size_t)b * CIN + kbase) * HW + n;
  v16h frag;
#pragma unroll
  for (int j = 0; j < 16; ++j) frag[j] = (_Float16)src[(size_t)j * HW];
  *(v16h*)(xt + ((((size_t)b * 8 + kk) * 256 + ntile) * 512 + lane * 16)) = frag;
}

// ---------------- WMMA A fragment: 16x32 f16, row-major [M][K] ----------------
__device__ inline v16h load_a_frag(const _Float16* __restrict__ A, int m0,
                                   int k0, int ldk, int lane) {
  const int mrow = m0 + (lane & 15);
  const int hi   = lane >> 4;
  const _Float16* ar = A + (size_t)mrow * ldk + k0;
  v16h af;
#pragma unroll
  for (int i = 0; i < 8; ++i) {
    const int ko = (hi << 4) + (i << 1);
    af[2 * i]     = ar[ko];
    af[2 * i + 1] = ar[ko + 1];
  }
  return af;
}

// ---------------- K1: qkv 1x1 conv as WMMA GEMM ----------------
// C[m,n] = sum_k Wqkv[m,k] * x[b,k,n];  M=768, K=256, N=4096 per batch.
// grid (32, 12, 8), block 256 (8 waves). Wave: 64(M) x 16(N), 4 accumulators.
__global__ void qkv_gemm_kernel(const _Float16* __restrict__ xt,
                                const _Float16* __restrict__ wq,
                                _Float16* __restrict__ qkv) {
  const int lane   = threadIdx.x & 31;
  const int wave   = threadIdx.x >> 5;
  const int b      = blockIdx.z;
  const int ntile  = blockIdx.x * 8 + wave;
  const int n0     = ntile * 16;
  const int m_base = blockIdx.y * 64;
  const int nlo    = lane & 15;
  const int hi     = lane >> 4;

  v8f acc[4] = {};
#pragma unroll
  for (int kk = 0; kk < 8; ++kk) {
    const _Float16* bp =
        xt + ((((size_t)b * 8 + kk) * 256 + ntile) * 512 + lane * 16);
    v16h bf = *(const v16h*)bp;
    if (kk < 7) __builtin_prefetch(bp + 256 * 512, 0, 1);
#pragma unroll
    for (int t = 0; t < 4; ++t) {
      v16h af = load_a_frag(wq, m_base + t * 16, kk * 32, CIN, lane);
      acc[t]  = __builtin_amdgcn_wmma_f32_16x16x32_f16(
          false, af, false, bf, (short)0, acc[t], false, false);
    }
  }
#pragma unroll
  for (int t = 0; t < 4; ++t) {
#pragma unroll
    for (int r = 0; r < 8; ++r) {
      const int m = m_base + t * 16 + r + (hi << 3);
      qkv[((size_t)b * TD + m) * HW + (n0 + nlo)] = (_Float16)acc[t][r];
    }
  }
}

// ---------------- K2: depthwise 5x5 + grouped 1x1 (8->8) ----------------
// grid (16 pixel-tiles of 16x16, 96 pw-groups, 8 batch), block 256.
__global__ void dwpw_kernel(const _Float16* __restrict__ qkv,
                            const float* __restrict__ wdw,
                            const float* __restrict__ wpw,
                            _Float16* __restrict__ agg) {
  __shared__ float smem[8][20][20];
  __shared__ float sdw[8][25];
  __shared__ float spw[8][8];

  const int tid = threadIdx.x;
  const int p   = blockIdx.y;          // pw group: channels 8p..8p+7
  const int b   = blockIdx.z;
  const int ty0 = (blockIdx.x >> 2) * 16;
  const int tx0 = (blockIdx.x & 3) * 16;

  const _Float16* src = qkv + ((size_t)b * TD + p * 8) * HW;

  for (int i = tid; i < 8 * 20 * 20; i += 256) {
    const int ch  = i / 400;
    const int rem = i % 400;
    const int yy  = rem / 20, xx = rem % 20;
    const int gy  = ty0 + yy - 2, gx = tx0 + xx - 2;
    float v = 0.0f;
    if (gy >= 0 && gy < WDIM && gx >= 0 && gx < WDIM)
      v = (float)src[(size_t)ch * HW + gy * WDIM + gx];
    smem[ch][yy][xx] = v;
  }
  if (tid < 200) sdw[tid / 25][tid % 25] = wdw[(size_t)(p * 8 + tid / 25) * 25 + tid % 25];
  if (tid < 64)  spw[tid / 8][tid % 8]   = wpw[(size_t)(p * 8 + tid / 8) * 8 + tid % 8];
  __syncthreads();

  const int ty = tid >> 4, tx = tid & 15;
  float dwv[8];
#pragma unroll
  for (int ch = 0; ch < 8; ++ch) {
    float a = 0.0f;
#pragma unroll
    for (int dy = 0; dy < 5; ++dy)
#pragma unroll
      for (int dx = 0; dx < 5; ++dx)
        a = fmaf(smem[ch][ty + dy][tx + dx], sdw[ch][dy * 5 + dx], a);
    dwv[ch] = a;
  }
  const int n = (ty0 + ty) * WDIM + (tx0 + tx);
#pragma unroll
  for (int co = 0; co < 8; ++co) {
    float a = 0.0f;
#pragma unroll
    for (int ci = 0; ci < 8; ++ci) a = fmaf(dwv[ci], spw[co][ci], a);
    agg[((size_t)b * TD + p * 8 + co) * HW + n] = (_Float16)a;
  }
}

// ---------------- K3: fused linear attention per (b, g) ----------------
// Phase 1: kv = relu(k)^T [v|1] reduced across 4096 pixels (shfl + LDS).
// Phase 2: out = relu(q) . kv, normalize, store f16 in fragment-major layout.
// grid (64, 8), block 256 (16 px/thread per phase).
__global__ void attn_kernel(const _Float16* __restrict__ qkv,
                            const _Float16* __restrict__ agg,
                            _Float16* __restrict__ attn_t) {
  __shared__ float red[72];
  const int tid = threadIdx.x;
  const int g   = blockIdx.x;
  const int b   = blockIdx.y;

  const _Float16* src = (g < 32)
      ? qkv + ((size_t)b * TD + 24 * g) * HW
      : agg + ((size_t)b * TD + 24 * (g - 32)) * HW;

  float kvl[8][9] = {};
#pragma unroll 2
  for (int it = 0; it < 16; ++it) {
    const int n = it * 256 + tid;
    float kf[8], vf[8];
#pragma unroll
    for (int d = 0; d < 8; ++d) kf[d] = fmaxf((float)src[(size_t)(8 + d) * HW + n], 0.0f);
#pragma unroll
    for (int e = 0; e < 8; ++e) vf[e] = (float)src[(size_t)(16 + e) * HW + n];
#pragma unroll
    for (int d = 0; d < 8; ++d) {
#pragma unroll
      for (int e = 0; e < 8; ++e) kvl[d][e] = fmaf(kf[d], vf[e], kvl[d][e]);
      kvl[d][8] += kf[d];
    }
  }

  if (tid < 72) red[tid] = 0.0f;
  __syncthreads();
#pragma unroll
  for (int d = 0; d < 8; ++d)
#pragma unroll
    for (int e = 0; e < 9; ++e) {
      float v = kvl[d][e];
      v += __shfl_xor(v, 16, 32);
      v += __shfl_xor(v, 8, 32);
      v += __shfl_xor(v, 4, 32);
      v += __shfl_xor(v, 2, 32);
      v += __shfl_xor(v, 1, 32);
      if ((tid & 31) == 0) atomicAdd(&red[d * 9 + e], v);  // ds_add_f32
    }
  __syncthreads();

  // destination tiling constants for channels c = g*8 + e
  const int kk      = g >> 2;            // c >> 5
  const int lanegrp = (g >> 1) & 1;      // (c&31) >> 4
  const int j0      = (g & 1) * 8;       // c & 15 base

#pragma unroll 2
  for (int it = 0; it < 16; ++it) {
    const int n = it * 256 + tid;
    float q[8];
#pragma unroll
    for (int d = 0; d < 8; ++d) q[d] = fmaxf((float)src[(size_t)d * HW + n], 0.0f);
    float acc[9];
#pragma unroll
    for (int e = 0; e < 9; ++e) {
      float a = 0.0f;
#pragma unroll
      for (int d = 0; d < 8; ++d) a = fmaf(q[d], red[d * 9 + e], a);
      acc[e] = a;
    }
    const float inv = 1.0f / (acc[8] + 1e-15f);
    v8h res;
#pragma unroll
    for (int e = 0; e < 8; ++e) res[e] = (_Float16)(acc[e] * inv);
    _Float16* dst = attn_t +
        ((((size_t)b * 16 + kk) * 256 + (n >> 4)) * 512 +
         (lanegrp * 16 + (n & 15)) * 16 + j0);
    *(v8h*)dst = res;
  }
}

// ---------------- K4: proj 1x1 conv (WMMA GEMM) + BN ----------------
// C[m,n] = sum_k Wproj[m,k]*attn[b,k,n]; M=256, K=512, N=4096 per batch.
// grid (32, 4, 8), block 256.
__global__ void proj_gemm_kernel(const _Float16* __restrict__ attn_t,
                                 const _Float16* __restrict__ wp,
                                 const float* __restrict__ scale,
                                 const float* __restrict__ shift,
                                 float* __restrict__ out) {
  const int lane   = threadIdx.x & 31;
  const int wave   = threadIdx.x >> 5;
  const int b      = blockIdx.z;
  const int ntile  = blockIdx.x * 8 + wave;
  const int n0     = ntile * 16;
  const int m_base = blockIdx.y * 64;
  const int nlo    = lane & 15;
  const int hi     = lane >> 4;

  v8f acc[4] = {};
#pragma unroll
  for (int kk = 0; kk < 16; ++kk) {
    const _Float16* bp =
        attn_t + ((((size_t)b * 16 + kk) * 256 + ntile) * 512 + lane * 16);
    v16h bf = *(const v16h*)bp;
    if (kk < 15) __builtin_prefetch(bp + 256 * 512, 0, 1);
#pragma unroll
    for (int t = 0; t < 4; ++t) {
      v16h af = load_a_frag(wp, m_base + t * 16, kk * 32, CMID, lane);
      acc[t]  = __builtin_amdgcn_wmma_f32_16x16x32_f16(
          false, af, false, bf, (short)0, acc[t], false, false);
    }
  }
#pragma unroll
  for (int t = 0; t < 4; ++t) {
#pragma unroll
    for (int r = 0; r < 8; ++r) {
      const int m = m_base + t * 16 + r + (hi << 3);
      out[((size_t)b * COUT + m) * HW + (n0 + nlo)] =
          acc[t][r] * scale[m] + shift[m];
    }
  }
}

// ---------------- launcher ----------------
extern "C" void kernel_launch(void* const* d_in, const int* in_sizes, int n_in,
                              void* d_out, int out_size, void* d_ws, size_t ws_size,
                              hipStream_t stream) {
  const float* x       = (const float*)d_in[0];
  const float* w_qkv   = (const float*)d_in[1];
  const float* w_dw    = (const float*)d_in[2];
  const float* w_pw    = (const float*)d_in[3];
  const float* w_proj  = (const float*)d_in[4];
  const float* g       = (const float*)d_in[5];
  const float* be      = (const float*)d_in[6];
  const float* mu      = (const float*)d_in[7];
  const float* va      = (const float*)d_in[8];
  float* out           = (float*)d_out;

  char* ws = (char*)d_ws;
  _Float16* wqkv16 = (_Float16*)(ws + OFF_WQKV);
  _Float16* wproj16= (_Float16*)(ws + OFF_WPROJ);
  float*    scale  = (float*)(ws + OFF_SCALE);
  float*    shift  = (float*)(ws + OFF_SHIFT);
  _Float16* xt     = (_Float16*)(ws + OFF_XT);
  _Float16* qkv    = (_Float16*)(ws + OFF_QKV);
  _Float16* agg    = (_Float16*)(ws + OFF_AGG);
  _Float16* attn   = (_Float16*)(ws + OFF_ATTN);

  prep_kernel<<<dim3(768), 256, 0, stream>>>(w_qkv, w_proj, g, be, mu, va,
                                             wqkv16, wproj16, scale, shift);
  xtile_kernel<<<dim3(32, 8, 8), 256, 0, stream>>>(x, xt);
  qkv_gemm_kernel<<<dim3(32, 12, 8), 256, 0, stream>>>(xt, wqkv16, qkv);
  dwpw_kernel<<<dim3(16, 96, 8), 256, 0, stream>>>(qkv, w_dw, w_pw, agg);
  attn_kernel<<<dim3(64, 8), 256, 0, stream>>>(qkv, agg, attn);
  proj_gemm_kernel<<<dim3(32, 4, 8), 256, 0, stream>>>(attn, wproj16, scale, shift, out);
}
